// Pointnet2Backbone_35785667510629
// MI455X (gfx1250) — matrix operand order repaired and checked by reading, compiled, not run on hardware
//
#include <hip/hip_runtime.h>
#include <hip/hip_bf16.h>
#include <stdint.h>

// ---------- vector types ----------
typedef __attribute__((ext_vector_type(16))) __bf16 v16bf;
typedef __attribute__((ext_vector_type(8)))  __bf16 v8bf;
typedef __attribute__((ext_vector_type(8)))  float  v8f;
typedef __attribute__((ext_vector_type(4)))  float  v4f;

__device__ __forceinline__ __bf16 f2bf(float f) {
  unsigned u = __builtin_bit_cast(unsigned, f);
  unsigned r = u + 0x7FFFu + ((u >> 16) & 1u);   // round-to-nearest-even
  return __builtin_bit_cast(__bf16, (unsigned short)(r >> 16));
}
__device__ __forceinline__ float bf2f(__bf16 h) {
  unsigned u = ((unsigned)__builtin_bit_cast(unsigned short, h)) << 16;
  return __builtin_bit_cast(float, u);
}

// fragment loaders: contiguous 16B segments matching the ISA VGPR layouts
__device__ __forceinline__ v16bf load_frag_a(const __bf16* __restrict__ row, int g) {
  v8bf x0 = *(const v8bf*)(row + 8 * g);        // k in [8g, 8g+8)
  v8bf x1 = *(const v8bf*)(row + 16 + 8 * g);   // k in [16+8g, +8)
  return __builtin_shufflevector(x0, x1, 0,1,2,3,4,5,6,7,8,9,10,11,12,13,14,15);
}
__device__ __forceinline__ v16bf load_frag_b(const __bf16* __restrict__ row, int g) {
  v8bf x0 = *(const v8bf*)(row + 8 * g);        // k in [8g, 8g+8)
  v8bf x1 = *(const v8bf*)(row + 16 + 8 * g);   // k in [16+8g, +8)
  return __builtin_shufflevector(x0, x1, 0,1,2,3,4,5,6,7,8,9,10,11,12,13,14,15);
}
#define WMMA_BF16(a, b, c) __builtin_amdgcn_wmma_f32_16x16x32_bf16(false, (a), false, (b), (short)0, (c), false, false)

// shared epilogue for one 16x16 sub-tile (lane holds 8 consecutive channels)
template<bool RELU, int OUTMODE>
__device__ __forceinline__ void epilogue_tile(
    v8f acc, int o0, int m, int b, void* __restrict__ Cv, long sC, int ldc,
    const __bf16* __restrict__ Res, long sR, int ldr,
    const float* __restrict__ bias, float scale)
{
  float out[8];
#pragma unroll
  for (int v = 0; v < 8; ++v) out[v] = acc[v] * scale;
  if (bias) {
    v4f bl0 = *(const v4f*)(bias + o0);
    v4f bl1 = *(const v4f*)(bias + o0 + 4);
#pragma unroll
    for (int v = 0; v < 4; ++v) { out[v] += bl0[v]; out[v + 4] += bl1[v]; }
  }
  if (Res) {
    v8bf rv = *(const v8bf*)(Res + (long)b * sR + (long)m * ldr + o0);
#pragma unroll
    for (int v = 0; v < 8; ++v) out[v] += bf2f(rv[v]);
  }
  if (RELU) {
#pragma unroll
    for (int v = 0; v < 8; ++v) out[v] = fmaxf(out[v], 0.f);
  }
  if (OUTMODE == 0) {            // bf16 point-major: one 16B store
    __bf16* C = (__bf16*)Cv + (long)b * sC;
    v8bf pv;
#pragma unroll
    for (int v = 0; v < 8; ++v) pv[v] = f2bf(out[v]);
    *(v8bf*)(C + (long)m * ldc + o0) = pv;
  } else if (OUTMODE == 1) {     // f32 point-major: two 16B stores
    float* C = (float*)Cv + (long)b * sC;
    v4f p0, p1;
#pragma unroll
    for (int v = 0; v < 4; ++v) { p0[v] = out[v]; p1[v] = out[v + 4]; }
    *(v4f*)(C + (long)m * ldc + o0)     = p0;
    *(v4f*)(C + (long)m * ldc + o0 + 4) = p1;
  } else if (OUTMODE == 2) {     // f32 channel-major (final reference layout)
    float* C = (float*)Cv + (long)b * sC;
#pragma unroll
    for (int v = 0; v < 8; ++v) C[(long)(o0 + v) * ldc + m] = out[v];
  } else {                       // bf16 channel-major (for V^T)
    __bf16* C = (__bf16*)Cv + (long)b * sC;
#pragma unroll
    for (int v = 0; v < 8; ++v) C[(long)(o0 + v) * ldc + m] = f2bf(out[v]);
  }
}

// =====================================================================
// 2x2 register-blocked GEMM (requires O%32==0 && M%32==0): per K-chunk
// 8 unconditional global_load_b128 feed 4 unconditional WMMAs (2:1).
// NO conditionals around WMMA: EXEC stays all-ones (ISA requirement).
//   Out[m][o] = act( scale * sum_k A[o][k]*B[m][k] + bias[o] + Res[m][o] )
// A: bf16 [O][lda] weights; B: bf16 [M][ldb] point-major; K%32==0.
// =====================================================================
template<bool RELU, int OUTMODE>
__global__ void __launch_bounds__(256) gemm_wmma_2x2(
    const __bf16* __restrict__ A, long sA, int lda,
    const __bf16* __restrict__ B, long sB, int ldb,
    void* __restrict__ Cv, long sC, int ldc,
    const __bf16* __restrict__ Res, long sR, int ldr,
    const float* __restrict__ bias,
    int O, int M, int K, float scale)
{
  const int lane = threadIdx.x & 31;
  const int wave = threadIdx.x >> 5;
  const int tiles_m = M >> 5;                  // 32-wide macro m-tiles
  const int tm = blockIdx.x * 8 + wave;        // uniform across wave
  const int to = blockIdx.y;
  if (tm >= tiles_m) return;                   // whole-wave uniform exit
  const int b = blockIdx.z;
  A += (long)b * sA;
  B += (long)b * sB;

  const int g = lane >> 4;
  const int r = lane & 15;
  const __bf16* __restrict__ A0 = A + (long)(to * 32 + r) * lda;
  const __bf16* __restrict__ A1 = A0 + 16L * lda;
  const __bf16* __restrict__ B0 = B + (long)(tm * 32 + r) * ldb;
  const __bf16* __restrict__ B1 = B0 + 16L * ldb;

  v8f acc00 = {}, acc01 = {}, acc10 = {}, acc11 = {};
  for (int kb = 0; kb < K; kb += 32) {
    v16bf af0 = load_frag_a(A0 + kb, g);
    v16bf af1 = load_frag_a(A1 + kb, g);
    v16bf bf0 = load_frag_b(B0 + kb, g);
    v16bf bf1 = load_frag_b(B1 + kb, g);
    acc00 = WMMA_BF16(af0, bf0, acc00);
    acc01 = WMMA_BF16(af0, bf1, acc01);
    acc10 = WMMA_BF16(af1, bf0, acc10);
    acc11 = WMMA_BF16(af1, bf1, acc11);
  }
  epilogue_tile<RELU, OUTMODE>(acc00, to*32 + 8*g,      tm*32 + r,      b, Cv, sC, ldc, Res, sR, ldr, bias, scale);
  epilogue_tile<RELU, OUTMODE>(acc01, to*32 + 8*g,      tm*32 + 16 + r, b, Cv, sC, ldc, Res, sR, ldr, bias, scale);
  epilogue_tile<RELU, OUTMODE>(acc10, to*32 + 16 + 8*g, tm*32 + r,      b, Cv, sC, ldc, Res, sR, ldr, bias, scale);
  epilogue_tile<RELU, OUTMODE>(acc11, to*32 + 16 + 8*g, tm*32 + 16 + r, b, Cv, sC, ldc, Res, sR, ldr, bias, scale);
}

// =====================================================================
// Single-tile GEMM for ragged O/M (the small 'enhance' GEMMs, O or M = 48).
// Same contract; one 16x16 tile per wave, 4 loads : 1 WMMA, fully pipelined.
// =====================================================================
template<bool RELU, int OUTMODE>
__global__ void __launch_bounds__(256) gemm_wmma_1x1(
    const __bf16* __restrict__ A, long sA, int lda,
    const __bf16* __restrict__ B, long sB, int ldb,
    void* __restrict__ Cv, long sC, int ldc,
    const __bf16* __restrict__ Res, long sR, int ldr,
    const float* __restrict__ bias,
    int O, int M, int K, float scale)
{
  const int lane = threadIdx.x & 31;
  const int wave = threadIdx.x >> 5;
  const int tiles_m = M >> 4;
  const int tm = blockIdx.x * 8 + wave;
  const int to = blockIdx.y;
  if (tm >= tiles_m) return;                   // whole-wave uniform exit
  const int b = blockIdx.z;
  A += (long)b * sA;
  B += (long)b * sB;
  const int g = lane >> 4;
  const int r = lane & 15;
  const __bf16* __restrict__ Arow = A + (long)(to * 16 + r) * lda;
  const __bf16* __restrict__ Brow = B + (long)(tm * 16 + r) * ldb;
  v8f acc = {};
  for (int kb = 0; kb < K; kb += 32) {
    v16bf af = load_frag_a(Arow + kb, g);
    v16bf bf = load_frag_b(Brow + kb, g);
    acc = WMMA_BF16(af, bf, acc);
  }
  epilogue_tile<RELU, OUTMODE>(acc, to*16 + 8*g, tm*16 + r, b, Cv, sC, ldc, Res, sR, ldr, bias, scale);
}

// =====================================================================
// Farthest point sampling: one block per batch; running min-distance
// array resident in LDS (80 KB of the 320 KB WGP LDS).
// =====================================================================
__global__ void __launch_bounds__(1024) fps_kernel(
    const float* __restrict__ xyz, int N, int npoint, int* __restrict__ inds)
{
  __shared__ float s_dist[20000];
  __shared__ float s_vmax[32];
  __shared__ int   s_imax[32];
  __shared__ int   s_cur;
  const int b = blockIdx.x;
  const float* P = xyz + (long)b * N * 3;
  int* out = inds + (long)b * npoint;
  const int tid = threadIdx.x, nt = blockDim.x;
  for (int i = tid; i < N; i += nt) s_dist[i] = 1e10f;
  if (tid == 0) { out[0] = 0; s_cur = 0; }
  __syncthreads();
  for (int step = 1; step < npoint; ++step) {
    const int last = s_cur;
    const float lx = P[last*3+0], ly = P[last*3+1], lz = P[last*3+2];
    float bestv = -1.f; int besti = 0x7fffffff;
    for (int i = tid; i < N; i += nt) {
      float dx = P[i*3+0]-lx, dy = P[i*3+1]-ly, dz = P[i*3+2]-lz;
      float d = dx*dx + dy*dy + dz*dz;
      d = fminf(s_dist[i], d);
      s_dist[i] = d;
      if (d > bestv || (d == bestv && i < besti)) { bestv = d; besti = i; }
    }
    for (int off = 16; off > 0; off >>= 1) {
      float ov = __shfl_down(bestv, off, 32);
      int   oi = __shfl_down(besti, off, 32);
      if (ov > bestv || (ov == bestv && oi < besti)) { bestv = ov; besti = oi; }
    }
    if ((tid & 31) == 0) { s_vmax[tid >> 5] = bestv; s_imax[tid >> 5] = besti; }
    __syncthreads();
    if (tid < 32) {
      int nw = nt >> 5;
      bestv = (tid < nw) ? s_vmax[tid] : -1.f;
      besti = (tid < nw) ? s_imax[tid] : 0x7fffffff;
      for (int off = 16; off > 0; off >>= 1) {
        float ov = __shfl_down(bestv, off, 32);
        int   oi = __shfl_down(besti, off, 32);
        if (ov > bestv || (ov == bestv && oi < besti)) { bestv = ov; besti = oi; }
      }
      if (tid == 0) { out[step] = besti; s_cur = besti; }
    }
    __syncthreads();
  }
}

__global__ void gather_xyz_kernel(const float* __restrict__ xyz,
                                  const int* __restrict__ inds,
                                  float* __restrict__ out, int N, int S)
{
  int t = blockIdx.x * blockDim.x + threadIdx.x;
  int b = blockIdx.y;
  if (t >= S * 3) return;
  int s = t / 3, c = t % 3;
  out[(long)b*S*3 + t] = xyz[(long)b*N*3 + (long)inds[(long)b*S + s]*3 + c];
}

__global__ void ball_query_kernel(const float* __restrict__ xyz,
                                  const float* __restrict__ new_xyz,
                                  int N, int S, float r2, int ns,
                                  int* __restrict__ idx)
{
  int t = blockIdx.x * blockDim.x + threadIdx.x;
  int b = blockIdx.y;
  if (t >= S) return;
  const float* P = xyz + (long)b*N*3;
  const float cx = new_xyz[(long)b*S*3 + t*3+0];
  const float cy = new_xyz[(long)b*S*3 + t*3+1];
  const float cz = new_xyz[(long)b*S*3 + t*3+2];
  int* o = idx + ((long)b*S + t) * ns;
  int cnt = 0, first = 0;
  for (int i = 0; i < N && cnt < ns; ++i) {
    float dx = P[i*3+0]-cx, dy = P[i*3+1]-cy, dz = P[i*3+2]-cz;
    if (dx*dx + dy*dy + dz*dz < r2) { if (cnt == 0) first = i; o[cnt++] = i; }
  }
  for (; cnt < ns; ++cnt) o[cnt] = first;
}

// grouped input, point-major bf16: X[m][c], c padded to C0p with zeros
__global__ void group_build_kernel(const float* __restrict__ xyz,
                                   const float* __restrict__ new_xyz,
                                   const int* __restrict__ idx,
                                   const __bf16* __restrict__ feats, // [N][Cf]
                                   int N, int S, int ns, int C0p, int Cf,
                                   float inv_r, __bf16* __restrict__ X)
{
  long t = (long)blockIdx.x * blockDim.x + threadIdx.x;
  long SN = (long)S * ns;
  if (t >= SN * C0p) return;
  long m = t / C0p;
  int  c = (int)(t % C0p);
  int  s = (int)(m / ns);
  int id = idx[m];
  float v = 0.f;
  if (c < 3) v = (xyz[(long)id*3 + c] - new_xyz[(long)s*3 + c]) * inv_r;
  else if (c - 3 < Cf) v = bf2f(feats[(long)id * Cf + (c - 3)]);
  X[t] = f2bf(v);
}

// max over ns samples: Y [S*ns][C] -> out [S][C]   (bf16, point-major)
__global__ void maxpool_kernel(const __bf16* __restrict__ Y, int C, int S, int ns,
                               __bf16* __restrict__ out)
{
  int t = blockIdx.x * blockDim.x + threadIdx.x;
  if (t >= S * C) return;
  int s = t / C, c = t % C;
  const __bf16* p = Y + (long)s * ns * C + c;
  float m = bf2f(p[0]);
  for (int n = 1; n < ns; ++n) m = fmaxf(m, bf2f(p[(long)n * C]));
  out[t] = f2bf(m);
}

// row softmax: scores f32 [Np][L] -> P bf16 [Np][Lp], pad cols zeroed
__global__ void softmax_row_kernel(const float* __restrict__ Sc,
                                   __bf16* __restrict__ P, int L, int Lp, int Np)
{
  int t = blockIdx.x * blockDim.x + threadIdx.x;
  int b = blockIdx.y;
  if (t >= Np) return;
  const float* row = Sc + (long)b*Np*L + (long)t*L;
  __bf16* prow = P + (long)b*Np*Lp + (long)t*Lp;
  float mx = -1e30f;
  for (int l = 0; l < L; ++l) mx = fmaxf(mx, row[l]);
  float sum = 0.f;
  float e[64];
  for (int l = 0; l < L; ++l) { e[l] = __expf(row[l] - mx); sum += e[l]; }
  float inv = 1.f / sum;
  for (int l = 0; l < L; ++l) prow[l] = f2bf(e[l] * inv);
  for (int l = L; l < Lp; ++l) prow[l] = f2bf(0.f);
}

__global__ void three_nn_kernel(const float* __restrict__ unk,
                                const float* __restrict__ knw,
                                int Nu, int Nk, int* __restrict__ idx,
                                float* __restrict__ w)
{
  int t = blockIdx.x * blockDim.x + threadIdx.x;
  int b = blockIdx.y;
  if (t >= Nu) return;
  const float* U = unk + (long)b*Nu*3;
  const float* Kp = knw + (long)b*Nk*3;
  float ux = U[t*3+0], uy = U[t*3+1], uz = U[t*3+2];
  float d0 = 1e30f, d1 = 1e30f, d2 = 1e30f; int i0 = 0, i1 = 0, i2 = 0;
  for (int i = 0; i < Nk; ++i) {
    float dx = Kp[i*3+0]-ux, dy = Kp[i*3+1]-uy, dz = Kp[i*3+2]-uz;
    float d = dx*dx + dy*dy + dz*dz;
    if      (d < d0) { d2=d1;i2=i1; d1=d0;i1=i0; d0=d;i0=i; }
    else if (d < d1) { d2=d1;i2=i1; d1=d;i1=i; }
    else if (d < d2) { d2=d;i2=i; }
  }
  float w0 = 1.f/(d0+1e-8f), w1 = 1.f/(d1+1e-8f), w2 = 1.f/(d2+1e-8f);
  float s = w0 + w1 + w2;
  long o = ((long)b*Nu + t) * 3;
  idx[o]=i0; idx[o+1]=i1; idx[o+2]=i2;
  w[o]=w0/s; w[o+1]=w1/s; w[o+2]=w2/s;
}

// interp + concat, point-major bf16: X[n][0..C)=f1, X[n][C..2C)=interp(f2)
__global__ void interp_concat_kernel(const __bf16* __restrict__ f1, // [Nu][C]
                                     const __bf16* __restrict__ f2, // [Nk][C]
                                     const int* __restrict__ idx,
                                     const float* __restrict__ w,
                                     int C, int Nu, int Nk, __bf16* __restrict__ X)
{
  int t = blockIdx.x * blockDim.x + threadIdx.x;
  int b = blockIdx.y;
  if (t >= C * Nu) return;
  int n = t / C, c = t % C;
  const __bf16* F1 = f1 + (long)b*Nu*C;
  const __bf16* F2 = f2 + (long)b*Nk*C;
  long o = ((long)b*Nu + n) * 3;
  float v = w[o]   * bf2f(F2[(long)idx[o]  *C + c])
          + w[o+1] * bf2f(F2[(long)idx[o+1]*C + c])
          + w[o+2] * bf2f(F2[(long)idx[o+2]*C + c]);
  __bf16* Xb = X + (long)b * (2L*C) * Nu;
  Xb[(long)n*(2*C) + c]     = F1[(long)n*C + c];
  Xb[(long)n*(2*C) + C + c] = f2bf(v);
}

// ---- weight / input conversion helpers ----
__global__ void conv_w_kernel(const float* __restrict__ src, __bf16* __restrict__ dst,
                              int O, int K, int Kp)
{
  int t = blockIdx.x * blockDim.x + threadIdx.x;
  if (t >= O * Kp) return;
  int o = t / Kp, k = t % Kp;
  dst[t] = f2bf(k < K ? src[(long)o*K + k] : 0.f);
}
__global__ void conv_wt_kernel(const float* __restrict__ src, __bf16* __restrict__ dst,
                               int O, int K)
{
  int t = blockIdx.x * blockDim.x + threadIdx.x;
  if (t >= O * K) return;
  int o = t / K, k = t % K;
  dst[t] = f2bf(src[(long)k*O + o]);
}
__global__ void conv_txt_kernel(const float* __restrict__ text,
                                const float* __restrict__ attr,
                                __bf16* __restrict__ dst)
{
  int t = blockIdx.x * blockDim.x + threadIdx.x;
  int b = blockIdx.y;
  if (t >= 48 * 768) return;
  int l = t / 768, c = t % 768;
  float v = (l < 32) ? text[(long)b*32*768 + (long)l*768 + c]
                     : attr[(long)b*16*768 + (long)(l-32)*768 + c];
  dst[(long)b*48*768 + t] = f2bf(v);
}
__global__ void zero_bf16_kernel(__bf16* __restrict__ p, long n)
{
  long t = (long)blockIdx.x * blockDim.x + threadIdx.x;
  if (t < n) p[t] = f2bf(0.f);
}

// ---------------- host side ----------------
#define CEIL(a,b) (((a)+(b)-1)/(b))
static inline int pad32(int k) { return (k + 31) & ~31; }

static inline void gemm(hipStream_t st, bool relu, int outmode,
                        const __bf16* A, long sA, int lda,
                        const __bf16* B, long sB, int ldb,
                        void* C, long sC, int ldc,
                        const __bf16* R, long sR, int ldr,
                        const float* bias, int O, int M, int K, float scale, int nb)
{
  dim3 block(256, 1, 1);
  if ((O % 32 == 0) && (M % 32 == 0)) {
    dim3 grid((unsigned)CEIL(M / 32, 8), (unsigned)(O / 32), (unsigned)nb);
#define G2(rl, om) gemm_wmma_2x2<rl, om><<<grid, block, 0, st>>>(A,sA,lda,B,sB,ldb,C,sC,ldc,R,sR,ldr,bias,O,M,K,scale)
    if      ( relu && outmode == 0) G2(true, 0);
    else if (!relu && outmode == 0) G2(false, 0);
    else                            G2(true, 2);   // final layer: f32 channel-major
#undef G2
  } else {
    dim3 grid((unsigned)CEIL(CEIL(M, 16), 8), (unsigned)CEIL(O, 16), (unsigned)nb);
#define G1(rl, om) gemm_wmma_1x1<rl, om><<<grid, block, 0, st>>>(A,sA,lda,B,sB,ldb,C,sC,ldc,R,sR,ldr,bias,O,M,K,scale)
    if      (!relu && outmode == 0) G1(false, 0);
    else if (!relu && outmode == 1) G1(false, 1);
    else if (!relu && outmode == 3) G1(false, 3);
    else                            G1(true, 0);
#undef G1
  }
}

extern "C" void kernel_launch(void* const* d_in, const int* in_sizes, int n_in,
                              void* d_out, int out_size, void* d_ws, size_t ws_size,
                              hipStream_t stream) {
  (void)in_sizes; (void)n_in; (void)out_size; (void)ws_size;
  const int B = 4;
  const float* pc   = (const float*)d_in[0];
  const float* text = (const float*)d_in[1];
  const float* attr = (const float*)d_in[2];
  int p = 3;
  const float *saW[4][3], *saB[4][3];
  for (int i = 0; i < 4; ++i)
    for (int l = 0; l < 3; ++l) { saW[i][l] = (const float*)d_in[p++]; saB[i][l] = (const float*)d_in[p++]; }
  const float *fpW[2][2], *fpB[2][2];
  for (int i = 0; i < 2; ++i)
    for (int l = 0; l < 2; ++l) { fpW[i][l] = (const float*)d_in[p++]; fpB[i][l] = (const float*)d_in[p++]; }
  const float* Wt = (const float*)d_in[p++]; const float* bt = (const float*)d_in[p++];
  const float* Wq = (const float*)d_in[p++]; const float* Wk = (const float*)d_in[p++];
  const float* Wv = (const float*)d_in[p++]; const float* Wo = (const float*)d_in[p++];
  const float* W1 = (const float*)d_in[p++]; const float* b1 = (const float*)d_in[p++];
  const float* W2 = (const float*)d_in[p++]; const float* b2 = (const float*)d_in[p++];

  // ---- workspace layout ----
  size_t cur = 0;
  auto alloc = [&](size_t bytes) { size_t o = cur; cur += (bytes + 255) & ~(size_t)255; return (char*)d_ws + o; };
  float* xyz1 = (float*)alloc(sizeof(float)*B*2048*3);
  float* xyz2 = (float*)alloc(sizeof(float)*B*1024*3);
  float* xyz3 = (float*)alloc(sizeof(float)*B*512*3);
  float* xyz4 = (float*)alloc(sizeof(float)*B*256*3);
  int*   inds = (int*)  alloc(sizeof(int)*B*2048);
  int*   ball = (int*)  alloc(sizeof(int)*B*2048*64);
  int*   nnI  = (int*)  alloc(sizeof(int)*B*1024*3);
  float* nnW  = (float*)alloc(sizeof(float)*B*1024*3);
  __bf16* sa1f = (__bf16*)alloc(2u*B*2048*128);
  __bf16* sa2f = (__bf16*)alloc(2u*B*1024*256);
  __bf16* sa3f = (__bf16*)alloc(2u*B*512*256);
  __bf16* sa4f = (__bf16*)alloc(2u*B*256*256);
  __bf16* txtIn = (__bf16*)alloc(2u*B*48*768);
  __bf16* txtE  = (__bf16*)alloc(2u*B*48*256);
  __bf16* qE    = (__bf16*)alloc(2u*B*256*256);
  __bf16* kE    = (__bf16*)alloc(2u*B*48*256);
  __bf16* vT    = (__bf16*)alloc(2u*B*256*64);   // [d][l], l padded 48->64
  float*  ScF   = (float*) alloc(4u*B*256*48);   // scores f32 [n][48]
  __bf16* Pb    = (__bf16*)alloc(2u*B*256*64);   // softmax [n][64]
  __bf16* ctxE  = (__bf16*)alloc(2u*B*256*256);
  __bf16* xE    = (__bf16*)alloc(2u*B*256*256);
  __bf16* hE    = (__bf16*)alloc(2u*B*256*256);
  __bf16* sa4e  = (__bf16*)alloc(2u*B*256*256);
  __bf16* fpX  = (__bf16*)alloc(2u*B*1024*512);
  __bf16* fpT  = (__bf16*)alloc(2u*B*1024*256);
  __bf16* fp0o = (__bf16*)alloc(2u*B*512*256);
  __bf16* saWb[4][3]; int saKp[4][3];
  const int sa_ch[4][4] = {{3,64,64,128},{131,128,128,256},{259,128,128,256},{259,128,128,256}};
  for (int i = 0; i < 4; ++i)
    for (int l = 0; l < 3; ++l) {
      saKp[i][l] = pad32(sa_ch[i][l]);
      saWb[i][l] = (__bf16*)alloc(2u * sa_ch[i][l+1] * saKp[i][l]);
    }
  const int fpO[2][2] = {{256,256},{256,288}};
  const int fpK[2][2] = {{512,256},{512,256}};
  __bf16* fpWb[2][2];
  for (int i = 0; i < 2; ++i)
    for (int l = 0; l < 2; ++l) fpWb[i][l] = (__bf16*)alloc(2u * fpO[i][l] * fpK[i][l]);
  __bf16* WtT = (__bf16*)alloc(2u*256*768);
  __bf16* WqT = (__bf16*)alloc(2u*256*256);
  __bf16* WkT = (__bf16*)alloc(2u*256*256);
  __bf16* WvT = (__bf16*)alloc(2u*256*256);
  __bf16* WoT = (__bf16*)alloc(2u*256*256);
  __bf16* W1T = (__bf16*)alloc(2u*256*256);
  __bf16* W2T = (__bf16*)alloc(2u*256*256);
  __bf16* BUFA = (__bf16*)alloc(2u * 16777216u);
  __bf16* BUFB = (__bf16*)alloc(2u * 16777216u);

  // ---- weight conversions (small, once per call) ----
  for (int i = 0; i < 4; ++i)
    for (int l = 0; l < 3; ++l) {
      int O = sa_ch[i][l+1], K = sa_ch[i][l], Kp = saKp[i][l];
      conv_w_kernel<<<CEIL(O*Kp,256), 256, 0, stream>>>(saW[i][l], saWb[i][l], O, K, Kp);
    }
  for (int i = 0; i < 2; ++i)
    for (int l = 0; l < 2; ++l)
      conv_w_kernel<<<CEIL(fpO[i][l]*fpK[i][l],256), 256, 0, stream>>>(fpW[i][l], fpWb[i][l], fpO[i][l], fpK[i][l], fpK[i][l]);
  conv_wt_kernel<<<CEIL(256*768,256), 256, 0, stream>>>(Wt, WtT, 256, 768);
  conv_wt_kernel<<<CEIL(256*256,256), 256, 0, stream>>>(Wq, WqT, 256, 256);
  conv_wt_kernel<<<CEIL(256*256,256), 256, 0, stream>>>(Wk, WkT, 256, 256);
  conv_wt_kernel<<<CEIL(256*256,256), 256, 0, stream>>>(Wv, WvT, 256, 256);
  conv_wt_kernel<<<CEIL(256*256,256), 256, 0, stream>>>(Wo, WoT, 256, 256);
  conv_wt_kernel<<<CEIL(256*256,256), 256, 0, stream>>>(W1, W1T, 256, 256);
  conv_wt_kernel<<<CEIL(256*256,256), 256, 0, stream>>>(W2, W2T, 256, 256);
  conv_txt_kernel<<<dim3(CEIL(48*768,256), B), 256, 0, stream>>>(text, attr, txtIn);
  zero_bf16_kernel<<<CEIL(B*256*64,256), 256, 0, stream>>>(vT, (long)B*256*64);

  // ---- SA stages (host loop over batches reuses the big ping-pong bufs) ----
  const int   sa_np[4] = {2048, 1024, 512, 256};
  const float sa_r[4]  = {0.2f, 0.4f, 0.8f, 1.2f};
  const int   sa_ns[4] = {64, 32, 16, 16};
  const float* src_xyz[4]  = {pc, xyz1, xyz2, xyz3};
  const int    src_N[4]    = {20000, 2048, 1024, 512};
  float*       dst_xyz[4]  = {xyz1, xyz2, xyz3, xyz4};
  const __bf16* src_feat[4] = {nullptr, sa1f, sa2f, sa3f};
  const int     src_Cf[4]   = {0, 128, 256, 256};
  __bf16*       dst_feat[4] = {sa1f, sa2f, sa3f, sa4f};

  for (int li = 0; li < 4; ++li) {
    const int N = src_N[li], S = sa_np[li], ns = sa_ns[li];
    const int C0p = saKp[li][0], Cf = src_Cf[li];
    const int c1 = sa_ch[li][1], c2 = sa_ch[li][2], c3 = sa_ch[li][3];
    const float r = sa_r[li];
    fps_kernel<<<B, 1024, 0, stream>>>(src_xyz[li], N, S, inds);
    gather_xyz_kernel<<<dim3(CEIL(S*3,256), B), 256, 0, stream>>>(src_xyz[li], inds, dst_xyz[li], N, S);
    ball_query_kernel<<<dim3(CEIL(S,256), B), 256, 0, stream>>>(src_xyz[li], dst_xyz[li], N, S, r*r, ns, ball);
    const long SN = (long)S * ns;
    for (int b = 0; b < B; ++b) {
      const __bf16* fsrc = src_feat[li] ? src_feat[li] + (long)b * N * Cf : nullptr;
      group_build_kernel<<<(unsigned)CEIL(SN*C0p, 256), 256, 0, stream>>>(
          src_xyz[li] + (long)b*N*3, dst_xyz[li] + (long)b*S*3, ball + (long)b*SN,
          fsrc, N, S, ns, C0p, Cf, 1.0f/r, BUFA);
      gemm(stream, true, 0, saWb[li][0], 0, C0p, BUFA, 0, C0p, BUFB, 0, c1,
           nullptr, 0, 0, saB[li][0], c1, (int)SN, C0p, 1.f, 1);
      gemm(stream, true, 0, saWb[li][1], 0, c1, BUFB, 0, c1, BUFA, 0, c2,
           nullptr, 0, 0, saB[li][1], c2, (int)SN, c1, 1.f, 1);
      gemm(stream, true, 0, saWb[li][2], 0, c2, BUFA, 0, c2, BUFB, 0, c3,
           nullptr, 0, 0, saB[li][2], c3, (int)SN, c2, 1.f, 1);
      maxpool_kernel<<<CEIL(S*c3,256), 256, 0, stream>>>(BUFB, c3, S, ns,
          dst_feat[li] + (long)b * S * c3);
    }
  }

  // ---- enhance: cross-attention + FFN (all point-major [pts][ch]) ----
  gemm(stream, false, 0, WtT, 0, 768, txtIn, 48*768, 768, txtE, 48*256, 256,
       nullptr, 0, 0, bt, 256, 48, 768, 1.f, B);
  gemm(stream, false, 0, WqT, 0, 256, sa4f, 256*256, 256, qE, 256*256, 256,
       nullptr, 0, 0, nullptr, 256, 256, 256, 1.f, B);
  gemm(stream, false, 0, WkT, 0, 256, txtE, 48*256, 256, kE, 48*256, 256,
       nullptr, 0, 0, nullptr, 256, 48, 256, 1.f, B);
  gemm(stream, false, 3, WvT, 0, 256, txtE, 48*256, 256, vT, 256*64, 64,
       nullptr, 0, 0, nullptr, 256, 48, 256, 1.f, B);      // -> vT [d][l]
  gemm(stream, false, 1, kE, 48*256, 256, qE, 256*256, 256, ScF, 256*48, 48,
       nullptr, 0, 0, nullptr, 48, 256, 256, 1.f/16.f, B);
  softmax_row_kernel<<<dim3(1, B), 256, 0, stream>>>(ScF, Pb, 48, 64, 256);
  gemm(stream, false, 0, vT, 256*64, 64, Pb, 256*64, 64, ctxE, 256*256, 256,
       nullptr, 0, 0, nullptr, 256, 256, 64, 1.f, B);
  gemm(stream, false, 0, WoT, 0, 256, ctxE, 256*256, 256, xE, 256*256, 256,
       sa4f, 256*256, 256, nullptr, 256, 256, 256, 1.f, B);
  gemm(stream, true, 0, W1T, 0, 256, xE, 256*256, 256, hE, 256*256, 256,
       nullptr, 0, 0, b1, 256, 256, 256, 1.f, B);
  gemm(stream, false, 0, W2T, 0, 256, hE, 256*256, 256, sa4e, 256*256, 256,
       xE, 256*256, 256, b2, 256, 256, 256, 1.f, B);

  // ---- FP module 0: sa3 (512) <- sa4 (256) ----
  three_nn_kernel<<<dim3(CEIL(512,256), B), 256, 0, stream>>>(xyz3, xyz4, 512, 256, nnI, nnW);
  interp_concat_kernel<<<dim3(CEIL(256*512,256), B), 256, 0, stream>>>(sa3f, sa4e, nnI, nnW, 256, 512, 256, fpX);
  gemm(stream, true, 0, fpWb[0][0], 0, 512, fpX, 512*512, 512, fpT, 512*256, 256,
       nullptr, 0, 0, fpB[0][0], 256, 512, 512, 1.f, B);
  gemm(stream, true, 0, fpWb[0][1], 0, 256, fpT, 512*256, 256, fp0o, 512*256, 256,
       nullptr, 0, 0, fpB[0][1], 256, 512, 256, 1.f, B);

  // ---- FP module 1: sa2 (1024) <- sa3 (512); final out f32 channel-major ----
  three_nn_kernel<<<dim3(CEIL(1024,256), B), 256, 0, stream>>>(xyz2, xyz3, 1024, 512, nnI, nnW);
  interp_concat_kernel<<<dim3(CEIL(256*1024,256), B), 256, 0, stream>>>(sa2f, fp0o, nnI, nnW, 256, 1024, 512, fpX);
  gemm(stream, true, 0, fpWb[1][0], 0, 512, fpX, 1024*512, 512, fpT, 1024*256, 256,
       nullptr, 0, 0, fpB[1][0], 256, 1024, 512, 1.f, B);
  gemm(stream, true, 2, fpWb[1][1], 0, 256, fpT, 1024*256, 256, (float*)d_out, 288*1024, 1024,
       nullptr, 0, 0, fpB[1][1], 288, 1024, 256, 1.f, B);
}
